// ROILayer_51350628991231
// MI455X (gfx1250) — compile-verified, matrix operand order checked by reading, not run on hardware
//
#include <hip/hip_runtime.h>
#include <hip/hip_bf16.h>
#include <stdint.h>

// ---------------------------------------------------------------------------
// MI455X (gfx1250) implementation.
// pos -> crop+bilinear(3->6, padded 8x8, bf16, channel-last)
//     -> grouped-conv as bf16 WMMA GEMM, LDS-staged via async Global->LDS,
//        double buffered (ASYNCcnt)          -> bf16 feat
//     -> per-group FC as bf16 WMMA GEMM      -> f32 out.
// ---------------------------------------------------------------------------

#define B_ 128
#define C_ 512
#define G_ 20

typedef __attribute__((ext_vector_type(16))) __bf16 bf16x16;
typedef __attribute__((ext_vector_type(8)))  float  f32x8;
typedef __attribute__((ext_vector_type(4)))  int    v4i;

union Frag { uint4 u[2]; bf16x16 v; };

#if __has_builtin(__builtin_amdgcn_global_load_async_to_lds_b128)
#define HAVE_ASYNC_LDS 1
// exact-or-convertible pointer casts for the async builtin
#define GAS1(p) ((__attribute__((address_space(1))) v4i*)(p))
#define LAS3(p) ((__attribute__((address_space(3))) v4i*)(p))
#else
#define HAVE_ASYNC_LDS 0
#endif

static __device__ __forceinline__ void wait_async0() {
#if HAVE_ASYNC_LDS
#if __has_builtin(__builtin_amdgcn_s_wait_asynccnt)
    __builtin_amdgcn_s_wait_asynccnt(0);
#else
    asm volatile("s_wait_asynccnt 0x0" ::: "memory");
#endif
#endif
}

// ---------------- landmark -> crop positions -------------------------------
__global__ void pos_kernel(const float* __restrict__ y,
                           int* __restrict__ cy, int* __restrict__ cx) {
    int t = blockIdx.x * blockDim.x + threadIdx.x;
    if (t >= B_ * G_) return;
    const int   I1[10] = {21, 18, 19, 41, 38, 49, 48, 51, 61, 56};
    const int   I2[10] = {22, 25, 24, 46, 43, 53, 54, 57, 63, 58};
    const float CF[10] = {-0.5f, -1.0f / 3.0f, 1.0f / 3.0f, 1.0f,
                          0.f, 0.f, 0.f, 0.f, 0.f, 0.5f};
    int b = t / G_, gg = t % G_;
    int i = gg >> 1, j = gg & 1;
    float ruler = fabsf(y[b * 136 + 39 * 2] - y[b * 136 + 42 * 2]);
    float off   = ruler * CF[i];
    int   idx   = j ? I2[i] : I1[i];
    float px = y[b * 136 + idx * 2 + 0] * (1.0f / 16.0f);
    float py = (y[b * 136 + idx * 2 + 1] + off) * (1.0f / 16.0f);
    int ix = (int)px;  ix = ix < 1 ? 1 : (ix > 12 ? 12 : ix);
    int iy = (int)py;  iy = iy < 1 ? 1 : (iy > 12 ? 12 : iy);
    cx[t] = ix;
    cy[t] = iy;
}

// ---------------- crop + bilinear upsample -> padded bf16 tiles ------------
// upT layout: [g][b][pix(8x8 padded)][cin]
__global__ void crop_upsample_kernel(const float* __restrict__ x,
                                     const int* __restrict__ cy,
                                     const int* __restrict__ cx,
                                     __bf16* __restrict__ upT) {
    int t = blockIdx.x * blockDim.x + threadIdx.x;
    int cin = t & (C_ - 1);
    int gb  = t >> 9;
    int g   = gb % G_;
    int b   = gb / G_;
    if (b >= B_) return;

    int sy = cy[b * G_ + g] - 1;
    int sx = cx[b * G_ + g] - 1;
    const float* xp = x + (((size_t)b * C_ + cin) * 14 + sy) * 14 + sx;

    float cr[3][3];
#pragma unroll
    for (int r = 0; r < 3; ++r)
#pragma unroll
        for (int s = 0; s < 3; ++s) cr[r][s] = xp[r * 14 + s];

    const float W[6][3] = {{1.f, 0.f, 0.f},      {0.75f, 0.25f, 0.f},
                           {0.25f, 0.75f, 0.f},  {0.f, 0.75f, 0.25f},
                           {0.f, 0.25f, 0.75f},  {0.f, 0.f, 1.f}};
    float tmp[6][3];
#pragma unroll
    for (int u = 0; u < 6; ++u)
#pragma unroll
        for (int s = 0; s < 3; ++s)
            tmp[u][s] = W[u][0] * cr[0][s] + W[u][1] * cr[1][s] + W[u][2] * cr[2][s];
    float up[6][6];
#pragma unroll
    for (int u = 0; u < 6; ++u)
#pragma unroll
        for (int v = 0; v < 6; ++v)
            up[u][v] = tmp[u][0] * W[v][0] + tmp[u][1] * W[v][1] + tmp[u][2] * W[v][2];

    __bf16* o = upT + ((size_t)(g * B_ + b) * 64) * C_ + cin;
#pragma unroll
    for (int row = 0; row < 8; ++row)
#pragma unroll
        for (int col = 0; col < 8; ++col) {
            float v = (row == 0 || row == 7 || col == 0 || col == 7)
                          ? 0.f : up[row - 1][col - 1];
            o[(size_t)(row * 8 + col) * C_] = (__bf16)v;
        }
}

// ---------------- weight repacks (f32 -> bf16) -----------------------------
__global__ void convw_prep_kernel(const float* __restrict__ w, __bf16* __restrict__ o) {
    size_t t = (size_t)blockIdx.x * blockDim.x + threadIdx.x;
    if (t >= (size_t)10240 * 4608) return;
    int    cin  = (int)(t % 512);
    size_t r    = t / 512;
    int    kidx = (int)(r % 9);
    size_t gm   = r / 9;
    o[t] = (__bf16)w[(gm * 512 + cin) * 9 + kidx];
}

__global__ void fcw_prep_kernel(const float* __restrict__ w, __bf16* __restrict__ o) {
    size_t t = (size_t)blockIdx.x * blockDim.x + threadIdx.x;
    if (t >= (size_t)G_ * 160 * 18432) return;
    int    k = (int)(t % 18432);
    size_t r = t / 18432;
    int    m = (int)(r % 160);
    int    g = (int)(r / 160);
    float  v = (m < 150) ? w[((size_t)g * 18432 + k) * 150 + m] : 0.0f;
    o[t] = (__bf16)v;
}

// ---------------- grouped conv: LDS-staged bf16 WMMA GEMM ------------------
// Per g: A[512 x 4608] * B[4608 x 4608], K ordered kidx*512+cin.
// Block: 256 thr, tile M=128 x N=64, K-stage 64, double-buffered async->LDS.
// Waves 4(M) x 2(N); each wave: 2x2 WMMA tiles.
#define KS   64
#define LDR  72   /* row stride in elements: 64 + 8 pad -> 144B, 16B aligned */

__global__ void __launch_bounds__(256)
conv_gemm_kernel(const __bf16* __restrict__ Aw, const __bf16* __restrict__ Bup,
                 const float* __restrict__ bias, __bf16* __restrict__ feat) {
    __shared__ __attribute__((aligned(16))) __bf16 sA[2][128 * LDR];
    __shared__ __attribute__((aligned(16))) __bf16 sB[2][64 * LDR];

    const int tid  = threadIdx.x;
    const int wave = tid >> 5;
    const int lane = tid & 31;
    const int half = lane >> 4;
    const int l16  = lane & 15;
    const int wm   = wave >> 1;          // 0..3  (M quarter, 32 rows)
    const int wn   = wave & 1;           // 0..1  (N half, 32 cols)

    const int bid = blockIdx.x;
    const int g   = bid / 288;           // 4 Mblocks * 72 Nblocks per group
    const int rem = bid % 288;
    const int mb  = rem / 72;
    const int nb  = rem % 72;

    // copy assignments: seg = 16B chunk within 128B K-row
    const int seg = tid & 7;
    const int r0  = tid >> 3;            // 0..31
    size_t gA[4], gB[2];
#pragma unroll
    for (int it = 0; it < 4; ++it) {
        int row = r0 + it * 32;          // 0..127
        gA[it] = (size_t)(g * 512 + mb * 128 + row) * 4608 + seg * 8;
    }
#pragma unroll
    for (int it = 0; it < 2; ++it) {
        int row = r0 + it * 32;          // 0..63
        int n = nb * 64 + row;
        int b = n / 36, p = n % 36;
        gB[it] = (size_t)(g * B_ + b) * 32768 +
                 (size_t)((p / 6) * 8 + (p % 6)) * 512 + seg * 8;
    }

    auto copy_stage = [&](int buf, int s) {
        const int k0   = s * KS;
        const int kidx = k0 >> 9;
        const int cin0 = k0 & 511;
        const size_t boff = (size_t)((kidx / 3) * 8 + (kidx % 3)) * 512 + cin0;
#pragma unroll
        for (int it = 0; it < 4; ++it) {
            int dst = (r0 + it * 32) * LDR + seg * 8;
#if HAVE_ASYNC_LDS
            __builtin_amdgcn_global_load_async_to_lds_b128(
                GAS1(Aw + gA[it] + k0), LAS3(&sA[buf][dst]), 0, 0);
#else
            *(uint4*)&sA[buf][dst] = *(const uint4*)(Aw + gA[it] + k0);
#endif
        }
#pragma unroll
        for (int it = 0; it < 2; ++it) {
            int dst = (r0 + it * 32) * LDR + seg * 8;
#if HAVE_ASYNC_LDS
            __builtin_amdgcn_global_load_async_to_lds_b128(
                GAS1(Bup + gB[it] + boff), LAS3(&sB[buf][dst]), 0, 0);
#else
            *(uint4*)&sB[buf][dst] = *(const uint4*)(Bup + gB[it] + boff);
#endif
        }
    };

    f32x8 acc[2][2] = {};

    copy_stage(0, 0);
#pragma unroll 1
    for (int s = 0; s < 72; ++s) {
        wait_async0();
        __syncthreads();
        if (s + 1 < 72) copy_stage((s + 1) & 1, s + 1);

        const __bf16* Ab = sA[s & 1];
        const __bf16* Bb = sB[s & 1];
#pragma unroll
        for (int sub = 0; sub < 2; ++sub) {
            const int e0 = sub * 32 + half * 8;
            Frag bf[2];
#pragma unroll
            for (int nt = 0; nt < 2; ++nt) {
                const __bf16* p = Bb + (wn * 32 + nt * 16 + l16) * LDR + e0;
                bf[nt].u[0] = *(const uint4*)p;
                bf[nt].u[1] = *(const uint4*)(p + 16);
            }
#pragma unroll
            for (int mt = 0; mt < 2; ++mt) {
                const __bf16* p = Ab + (wm * 32 + mt * 16 + l16) * LDR + e0;
                Frag af;
                af.u[0] = *(const uint4*)p;
                af.u[1] = *(const uint4*)(p + 16);
                acc[mt][0] = __builtin_amdgcn_wmma_f32_16x16x32_bf16(
                    false, af.v, false, bf[0].v, (short)0, acc[mt][0], false, false);
                acc[mt][1] = __builtin_amdgcn_wmma_f32_16x16x32_bf16(
                    false, af.v, false, bf[1].v, (short)0, acc[mt][1], false, false);
            }
        }
    }

    // bias + relu + store bf16 feat[b][g][c*36+p]
#pragma unroll
    for (int nt = 0; nt < 2; ++nt) {
        int n = nb * 64 + wn * 32 + nt * 16 + l16;
        int b = n / 36, p = n % 36;
        size_t fb = ((size_t)b * G_ + g) * 18432 + p;
#pragma unroll
        for (int mt = 0; mt < 2; ++mt) {
#pragma unroll
            for (int r = 0; r < 8; ++r) {
                int m = mb * 128 + wm * 32 + mt * 16 + r + half * 8;
                float v = acc[mt][nt][r] + bias[g * 512 + m];
                v = v > 0.f ? v : 0.f;
                feat[fb + (size_t)m * 36] = (__bf16)v;
            }
        }
    }
}

// ---------------- per-group FC as bf16 WMMA GEMM ---------------------------
__global__ void __launch_bounds__(256)
fc_gemm_kernel(const __bf16* __restrict__ Aw, const __bf16* __restrict__ feat,
               const float* __restrict__ fcb, float* __restrict__ out) {
    int wu   = blockIdx.x * 8 + (threadIdx.x >> 5);
    int lane = threadIdx.x & 31;
    int half = lane >> 4;
    int l16  = lane & 15;

    int g     = wu / 20;
    int rem   = wu % 20;
    int mBase = (rem / 2) * 16;
    int nBase = (rem % 2) * 64;

    size_t aRow = (size_t)(g * 160 + mBase + l16) * 18432 + half * 8;
    size_t bRow[4];
#pragma unroll
    for (int t = 0; t < 4; ++t)
        bRow[t] = ((size_t)(nBase + t * 16 + l16) * G_ + g) * 18432 + half * 8;

    f32x8 acc[4] = {};

#pragma unroll 1
    for (int k = 0; k < 18432; k += 32) {
        Frag a;
        const uint4* pa = (const uint4*)(Aw + aRow + k);
        a.u[0] = pa[0];
        a.u[1] = pa[2];
#pragma unroll
        for (int t = 0; t < 4; ++t) {
            Frag bb;
            const uint4* pb = (const uint4*)(feat + bRow[t] + k);
            bb.u[0] = pb[0];
            bb.u[1] = pb[2];
            acc[t] = __builtin_amdgcn_wmma_f32_16x16x32_bf16(
                false, a.v, false, bb.v, (short)0, acc[t], false, false);
        }
    }

#pragma unroll
    for (int t = 0; t < 4; ++t) {
        int b = nBase + t * 16 + l16;
#pragma unroll
        for (int r = 0; r < 8; ++r) {
            int o = mBase + r + half * 8;
            if (o < 150) {
                float v = acc[t][r] + fcb[g * 150 + o];
                v = v > 0.f ? v : 0.f;
                out[(size_t)b * 3000 + g * 150 + o] = v;
            }
        }
    }
}

// ---------------------------------------------------------------------------
extern "C" void kernel_launch(void* const* d_in, const int* in_sizes, int n_in,
                              void* d_out, int out_size, void* d_ws, size_t ws_size,
                              hipStream_t stream) {
    const float* x      = (const float*)d_in[0];
    const float* y      = (const float*)d_in[1];
    const float* conv_w = (const float*)d_in[2];
    const float* conv_b = (const float*)d_in[3];
    const float* fc_w   = (const float*)d_in[4];
    const float* fc_b   = (const float*)d_in[5];
    float*       out    = (float*)d_out;

    char* ws = (char*)d_ws;
    int*    cy    = (int*)(ws + 0);
    int*    cx    = (int*)(ws + 10240);
    __bf16* upT   = (__bf16*)(ws + 20480);            // 167,772,160 B
    __bf16* convA = (__bf16*)(ws + 167792640ull);     //  94,371,840 B
    __bf16* feat  = (__bf16*)(ws + 262164480ull);     //  94,371,840 B
    __bf16* fcA   = (__bf16*)(ws + 356536320ull);     // 117,964,800 B
    // total ~474.5 MB of d_ws

    pos_kernel<<<10, 256, 0, stream>>>(y, cy, cx);
    convw_prep_kernel<<<184320, 256, 0, stream>>>(conv_w, convA);
    fcw_prep_kernel<<<230400, 256, 0, stream>>>(fc_w, fcA);
    crop_upsample_kernel<<<5120, 256, 0, stream>>>(x, cy, cx, upT);
    conv_gemm_kernel<<<5760, 256, 0, stream>>>(convA, upT, conv_b, feat);
    fc_gemm_kernel<<<50, 256, 0, stream>>>(fcA, feat, fc_b, out);
}
